// CNN_6708738916816
// MI455X (gfx1250) — compile-verified, matrix operand order checked by reading, not run on hardware
//
#include <hip/hip_runtime.h>
#include <hip/hip_bf16.h>
#include <cstdint>
#include <cstddef>

typedef __bf16 bf16_t;
typedef __attribute__((ext_vector_type(16))) __bf16 v16bf;
typedef __attribute__((ext_vector_type(8)))  __bf16 v8bf;
typedef __attribute__((ext_vector_type(8)))  float  v8f;

union FragAB {
    v16bf v;
    v8bf  h[2];
};

__device__ __forceinline__ void wait_async0() {
#if __has_builtin(__builtin_amdgcn_s_wait_asynccnt)
    __builtin_amdgcn_s_wait_asynccnt(0);
#else
    asm volatile("s_wait_asynccnt 0x0" ::: "memory");
#endif
}

// ---------------- degree / normalization ----------------

__global__ __launch_bounds__(256) void k_init_deg(float* deg, int n) {
    int i = blockIdx.x * blockDim.x + threadIdx.x;
    if (i < n) deg[i] = 1.0f;   // self-loop contributes 1 to every node's degree
}

__global__ __launch_bounds__(256) void k_count_dst(const long long* __restrict__ dst,
                                                   float* deg, int e) {
    int i = blockIdx.x * blockDim.x + threadIdx.x;
    if (i < e) atomicAdd(&deg[(int)dst[i]], 1.0f);
}

__global__ __launch_bounds__(256) void k_rsqrt(float* deg, int n) {
    int i = blockIdx.x * blockDim.x + threadIdx.x;
    if (i < n) deg[i] = rsqrtf(deg[i]);
}

// ---------------- f32 -> bf16 convert (streaming) ----------------

__global__ __launch_bounds__(256) void k_cvt_bf16(const float* __restrict__ in,
                                                  bf16_t* __restrict__ out,
                                                  long long n) {
    long long i = (long long)(blockIdx.x * blockDim.x + threadIdx.x) * 4;
    long long stride = (long long)gridDim.x * blockDim.x * 4;
    for (; i < n; i += stride) {
        float4 v = *(const float4*)(in + i);
        out[i + 0] = (bf16_t)v.x;
        out[i + 1] = (bf16_t)v.y;
        out[i + 2] = (bf16_t)v.z;
        out[i + 3] = (bf16_t)v.w;
    }
}

// ---------------- WMMA GEMM: C[M,N] = A[M,K] * B[N,K]^T (+bias) ----------------
// Block: 256 threads = 8 wave32, arranged 2 (M) x 4 (N); block tile 128x128.
// K-step 32. A/B 128x32 bf16 tiles are double-buffered in LDS (rows padded to
// 40 elems = 20 banks so 16-row ds_load_b128 fragment reads are conflict-free).
// Tiles are filled with GLOBAL_LOAD_ASYNC_TO_LDS_B128 (memory -> LDS direct,
// tracked by ASYNCcnt): async loads for step k+1 are issued before step k's
// compute, then drained with s_wait_asynccnt 0 + workgroup barrier.
// Each wave computes 4x2 = 8 accumulator tiles of 16x16 (8 WMMAs per K-step).

#define BK 32
#define LDS_STRIDE 40   // 32 data + 8 pad bf16 elements per row

__global__ __launch_bounds__(256) void k_gemm_xwT(const bf16_t* __restrict__ A,
                                                  const bf16_t* __restrict__ B,
                                                  const float* __restrict__ bias,
                                                  float* __restrict__ C,
                                                  int M, int Ncols, int K) {
    __shared__ bf16_t ldsA[2][128 * LDS_STRIDE];
    __shared__ bf16_t ldsB[2][128 * LDS_STRIDE];

    const int tid   = threadIdx.x;
    const int lane  = tid & 31;
    const int wave  = tid >> 5;
    const int waveM = wave >> 2;          // 0..1
    const int waveN = wave & 3;           // 0..3
    const int half  = lane >> 4;          // 0 or 1
    const int l16   = lane & 15;

    const int mblk = blockIdx.y * 128;
    const int nblk = blockIdx.x * 128;

    // Cooperative staging: 512 16B-chunks per tile / 256 threads = 2 chunks each.
    const int srow = tid >> 2;            // 0..63 (second chunk at srow+64)
    const int scol = (tid & 3) * 8;       // element offset 0/8/16/24
    const bf16_t* gA0 = A + (size_t)(mblk + srow) * K + scol;
    const bf16_t* gB0 = B + (size_t)(nblk + srow) * K + scol;
    const size_t  gskip = (size_t)64 * K;

    // Byte addresses inside the wave's LDS allocation (low 32 bits of the
    // generic shared-space pointer; HW adds LDS_BASE per the async-LDS rules).
    const unsigned bufBytes = 128u * LDS_STRIDE * 2u;  // 10240 B per buffer
    const unsigned lA0 = (unsigned)(uintptr_t)&ldsA[0][srow * LDS_STRIDE + scol];
    const unsigned lB0 = (unsigned)(uintptr_t)&ldsB[0][srow * LDS_STRIDE + scol];
    const unsigned lrowSkip = 64u * LDS_STRIDE * 2u;   // +64 rows, in bytes

    auto async_fetch = [&](int k, int buf) {
        const unsigned bo = buf ? bufBytes : 0u;
        unsigned long long a0 = (unsigned long long)(uintptr_t)(gA0 + k);
        unsigned long long a1 = (unsigned long long)(uintptr_t)(gA0 + gskip + k);
        unsigned long long b0 = (unsigned long long)(uintptr_t)(gB0 + k);
        unsigned long long b1 = (unsigned long long)(uintptr_t)(gB0 + gskip + k);
        asm volatile("global_load_async_to_lds_b128 %0, %1, off"
                     :: "v"(lA0 + bo), "v"(a0) : "memory");
        asm volatile("global_load_async_to_lds_b128 %0, %1, off"
                     :: "v"(lA0 + bo + lrowSkip), "v"(a1) : "memory");
        asm volatile("global_load_async_to_lds_b128 %0, %1, off"
                     :: "v"(lB0 + bo), "v"(b0) : "memory");
        asm volatile("global_load_async_to_lds_b128 %0, %1, off"
                     :: "v"(lB0 + bo + lrowSkip), "v"(b1) : "memory");
    };

    const v8f zero = {0.f, 0.f, 0.f, 0.f, 0.f, 0.f, 0.f, 0.f};
    v8f acc[4][2];
#pragma unroll
    for (int t = 0; t < 4; ++t)
#pragma unroll
        for (int u = 0; u < 2; ++u) acc[t][u] = zero;

    // Fragment read offsets in LDS (CDNA5 16-bit A/B layout: lane half h reads
    // K-chunks [h*8, h*8+8) and [16+h*8, 16+h*8+8)).
    const int aoff = (waveM * 64 + l16) * LDS_STRIDE + half * 8;
    const int boff = (waveN * 32 + l16) * LDS_STRIDE + half * 8;

    const int nk = K / BK;
    async_fetch(0, 0);
    wait_async0();
    __syncthreads();

    for (int ks = 0; ks < nk; ++ks) {
        const int cur = ks & 1;
        if (ks + 1 < nk) async_fetch((ks + 1) * BK, cur ^ 1);

        FragAB a[4], b[2];
#pragma unroll
        for (int t = 0; t < 4; ++t) {
            const bf16_t* p = &ldsA[cur][aoff + t * 16 * LDS_STRIDE];
            a[t].h[0] = *(const v8bf*)p;
            a[t].h[1] = *(const v8bf*)(p + 16);
        }
#pragma unroll
        for (int u = 0; u < 2; ++u) {
            const bf16_t* p = &ldsB[cur][boff + u * 16 * LDS_STRIDE];
            b[u].h[0] = *(const v8bf*)p;
            b[u].h[1] = *(const v8bf*)(p + 16);
        }
#pragma unroll
        for (int t = 0; t < 4; ++t)
#pragma unroll
            for (int u = 0; u < 2; ++u)
                acc[t][u] = __builtin_amdgcn_wmma_f32_16x16x32_bf16(
                    false, a[t].v, false, b[u].v, (short)0, acc[t][u], false, false);

        wait_async0();        // next buffer fully landed in LDS
        __syncthreads();      // ..for every wave; reads of 'cur' also retired
    }

    // D-matrix layout: VGPR r -> M = half*8 + r, N = lane&15 (per 16x16 tile).
#pragma unroll
    for (int t = 0; t < 4; ++t) {
        const int mb = mblk + waveM * 64 + t * 16 + half * 8;
#pragma unroll
        for (int u = 0; u < 2; ++u) {
            const int n = nblk + waveN * 32 + u * 16 + l16;
            const float bv = bias ? bias[n] : 0.0f;
#pragma unroll
            for (int r = 0; r < 8; ++r)
                C[(size_t)(mb + r) * Ncols + n] = acc[t][u][r] + bv;
        }
    }
}

// ---------------- aggregation ----------------

// H[v,:] = dinv[v]^2 * XW[v,:] + b[:]   (self-loop term + bias)
__global__ __launch_bounds__(256) void k_self_bias(const float* __restrict__ xw,
                                                   const float* __restrict__ dinv,
                                                   const float* __restrict__ b,
                                                   float* __restrict__ h, int F) {
    const int v = blockIdx.x;
    const float w = dinv[v] * dinv[v];
    const float* xr = xw + (size_t)v * F;
    float* hr = h + (size_t)v * F;
    for (int f = threadIdx.x; f < F; f += blockDim.x)
        hr[f] = xr[f] * w + b[f];
}

// H[dst,:] += dinv[src]*dinv[dst] * XW[src,:]   (one block per edge)
__global__ __launch_bounds__(256) void k_edge_agg(const long long* __restrict__ src,
                                                  const long long* __restrict__ dst,
                                                  const float* __restrict__ dinv,
                                                  const float* __restrict__ xw,
                                                  float* __restrict__ h, int F) {
    const int e = blockIdx.x;
    const int s = (int)src[e];
    const int d = (int)dst[e];
    const float w = dinv[s] * dinv[d];
    const float* xr = xw + (size_t)s * F;
    float* hr = h + (size_t)d * F;
    for (int f = threadIdx.x; f < F; f += blockDim.x)
        atomicAdd(&hr[f], xr[f] * w);
}

__global__ __launch_bounds__(256) void k_tanh(float* __restrict__ p, long long n) {
    long long i = blockIdx.x * blockDim.x + threadIdx.x;
    long long stride = (long long)gridDim.x * blockDim.x;
    for (; i < n; i += stride) p[i] = tanhf(p[i]);
}

// ---------------- launcher ----------------

extern "C" void kernel_launch(void* const* d_in, const int* in_sizes, int n_in,
                              void* d_out, int out_size, void* d_ws, size_t ws_size,
                              hipStream_t stream) {
    const float*     x   = (const float*)d_in[0];
    const long long* ei  = (const long long*)d_in[1];  // [2][E] int64
    const float* W1 = (const float*)d_in[2]; const float* b1 = (const float*)d_in[3];
    const float* W2 = (const float*)d_in[4]; const float* b2 = (const float*)d_in[5];
    const float* W3 = (const float*)d_in[6]; const float* b3 = (const float*)d_in[7];
    const float* Wl = (const float*)d_in[8]; const float* bl = (const float*)d_in[9];

    const int H1 = in_sizes[3], H2 = in_sizes[5], H3 = in_sizes[7], FOUT = in_sizes[9];
    const int FIN = in_sizes[2] / H1;
    const int Nn  = in_sizes[0] / FIN;
    const int E   = in_sizes[1] / 2;

    size_t hmax = (size_t)H1;
    if ((size_t)H2 > hmax) hmax = H2;
    if ((size_t)H3 > hmax) hmax = H3;
    if ((size_t)FIN > hmax) hmax = FIN;
    if ((size_t)FOUT > hmax) hmax = FOUT;

    size_t wmax = (size_t)H1 * FIN;
    if ((size_t)H2 * H1 > wmax) wmax = (size_t)H2 * H1;
    if ((size_t)H3 * H2 > wmax) wmax = (size_t)H3 * H2;
    if ((size_t)FOUT * H3 > wmax) wmax = (size_t)FOUT * H3;

    char* ws = (char*)d_ws;
    size_t off = 0;
    auto carve = [&](size_t bytes) -> char* {
        char* p = ws + off;
        off = (off + bytes + 255) & ~(size_t)255;
        return p;
    };
    float*  dinv = (float*)carve((size_t)Nn * 4);
    bf16_t* Abf  = (bf16_t*)carve((size_t)Nn * hmax * 2);
    bf16_t* Wbf  = (bf16_t*)carve(wmax * 2);
    float*  XW   = (float*)carve((size_t)Nn * hmax * 4);
    float*  Hbuf = (float*)carve((size_t)Nn * hmax * 4);
    (void)ws_size; (void)n_in; (void)out_size;

    const long long* srcIdx = ei;
    const long long* dstIdx = ei + E;

    // --- symmetric normalization coefficients ---
    k_init_deg<<<(Nn + 255) / 256, 256, 0, stream>>>(dinv, Nn);
    k_count_dst<<<(E + 255) / 256, 256, 0, stream>>>(dstIdx, dinv, E);
    k_rsqrt<<<(Nn + 255) / 256, 256, 0, stream>>>(dinv, Nn);

    auto cvt = [&](const float* in, bf16_t* outp, long long n) {
        long long groups = n / 4;
        int blocks = (int)((groups + 255) / 256);
        if (blocks > 4096) blocks = 4096;
        k_cvt_bf16<<<blocks, 256, 0, stream>>>(in, outp, n);
    };

    auto layer = [&](const float* in, const float* W, const float* b,
                     int K, int F, bool last, float* finalOut) {
        cvt(in, Abf, (long long)Nn * K);
        cvt(W,  Wbf, (long long)F * K);
        dim3 grid(F / 128, Nn / 128);
        k_gemm_xwT<<<grid, 256, 0, stream>>>(Abf, Wbf, last ? b : nullptr,
                                             last ? finalOut : XW, Nn, F, K);
        if (!last) {
            k_self_bias<<<Nn, 256, 0, stream>>>(XW, dinv, b, Hbuf, F);
            k_edge_agg<<<E, 256, 0, stream>>>(srcIdx, dstIdx, dinv, XW, Hbuf, F);
            long long n = (long long)Nn * F;
            int blocks = (int)((n + 255) / 256);
            if (blocks > 4096) blocks = 4096;
            k_tanh<<<blocks, 256, 0, stream>>>(Hbuf, n);
        }
    };

    layer(x,    W1, b1, FIN, H1,  false, nullptr);
    layer(Hbuf, W2, b2, H1,  H2,  false, nullptr);
    layer(Hbuf, W3, b3, H2,  H3,  false, nullptr);
    layer(Hbuf, Wl, bl, H3,  FOUT, true, (float*)d_out);
}
